// MultiheadSelfAttention_27247272526274
// MI455X (gfx1250) — compile-verified, matrix operand order checked by reading, not run on hardware
//
#include <hip/hip_runtime.h>

// ---------------------------------------------------------------------------
// MultiheadSelfAttention for MI455X (gfx1250): bf16 WMMA + TDM staging.
// B=4, S=2048, D=1024, H=16, dk=64.
// ---------------------------------------------------------------------------

typedef unsigned short u16;
typedef __attribute__((ext_vector_type(16))) __bf16        v16bf;
typedef __attribute__((ext_vector_type(8)))  float         v8f;
typedef __attribute__((ext_vector_type(4)))  unsigned int  v4u;
typedef __attribute__((ext_vector_type(8)))  int           v8i;
typedef __attribute__((ext_vector_type(4)))  int           v4i;

#define B_  4
#define S_  2048
#define D_  1024
#define H_  16
#define DK_ 64
#define M_  (B_ * S_)   // 8192 rows of x

// ---- Tensor Data Mover availability (device pass only) --------------------
#if defined(__gfx1250__) && __has_builtin(__builtin_amdgcn_tensor_load_to_lds)
#define USE_TDM 1
#else
#define USE_TDM 0
#endif

#if defined(__clang_major__) && (__clang_major__ >= 23)
#define TDM_SIX_ARGS 1
#else
#define TDM_SIX_ARGS 0
#endif

#if USE_TDM
#if __has_builtin(__builtin_amdgcn_s_wait_tensorcnt)
#define TDM_WAIT(n) __builtin_amdgcn_s_wait_tensorcnt(n)
#else
#define TDM_WAIT(n) asm volatile("s_wait_tensorcnt " #n ::: "memory")
#endif

// Issue one 2-D TDM tile load: tile_w x tile_h elements (2 bytes each) from a
// row-major tensor with row stride `row_stride` (elems) into LDS at lds_off.
// D# encoding per CDNA5 ISA 8.3/8.4: group0 {count=1, lds_addr, global_addr,
// type=2}; group1 {data_size=2B, tensor_dim0/1, tile_dim0/1, dim0_stride}.
__device__ __forceinline__ void tdm_load_2d(const void* gp, unsigned lds_off,
                                            int tile_w, int tile_h,
                                            int row_stride, int tensor_h) {
    unsigned long long ga = (unsigned long long)gp;
    v4u g0 = { 1u,                                    // count=1 (valid D#)
               lds_off,                               // lds_addr
               (unsigned)ga,                          // global_addr[31:0]
               ((unsigned)(ga >> 32) & 0x01ffffffu)   // global_addr[56:32]
                   | (2u << 30) };                    // type=2 ("image")
    v8i g1 = { (int)(1u << 16),                                        // data_size=1 -> 2B
               (int)(((unsigned)row_stride & 0xffffu) << 16),          // tensor_dim0 lo16
               (int)((((unsigned)row_stride >> 16) & 0xffffu)          // tensor_dim0 hi16
                     | (((unsigned)tensor_h & 0xffffu) << 16)),        // tensor_dim1 lo16
               (int)((((unsigned)tensor_h >> 16) & 0xffffu)            // tensor_dim1 hi16
                     | (((unsigned)tile_w & 0xffffu) << 16)),          // tile_dim0
               (int)((unsigned)tile_h & 0xffffu),                      // tile_dim1
               row_stride,                                             // dim0_stride lo32
               0, 0 };
    v4i z4 = { 0, 0, 0, 0 };
#if TDM_SIX_ARGS
    v8i z8 = { 0, 0, 0, 0, 0, 0, 0, 0 };
    __builtin_amdgcn_tensor_load_to_lds(g0, g1, z4, z4, z8, 0);
#else
    __builtin_amdgcn_tensor_load_to_lds(g0, g1, z4, z4, 0);
#endif
}
#endif // USE_TDM

union Frag {
    uint4 q[2];
    v16bf v;
};

__device__ __forceinline__ u16 f2bf(float f) {
    union { float f; unsigned u; } cv; cv.f = f;
    unsigned u = cv.u;
    u += 0x7fffu + ((u >> 16) & 1u);       // round-to-nearest-even
    return (u16)(u >> 16);
}

// Load a 16x32 bf16 fragment (A layout; B loads identically from a transposed
// tile). base points at row 0 of the 16-row group; stride is in bf16 elems.
// Lane L holds row L%16; elems e<8 -> K = kkoff + 8*(L/16) + e,
//                        e>=8 -> K = kkoff + 16 + 8*(L/16) + (e-8).
__device__ __forceinline__ Frag load_frag(const u16* base, int stride, int kkoff, int lane) {
    int row  = lane & 15;
    int half = lane >> 4;
    const u16* p = base + row * stride + kkoff + 8 * half;
    Frag f;
    f.q[0] = *reinterpret_cast<const uint4*>(p);
    f.q[1] = *reinterpret_cast<const uint4*>(p + 16);
    return f;
}

__device__ __forceinline__ v8f wmma_bf16(Frag a, Frag b, v8f c) {
    return __builtin_amdgcn_wmma_f32_16x16x32_bf16(
        /*neg_a=*/false, a.v, /*neg_b=*/false, b.v,
        /*c_mod=*/(short)0, c, /*reuse_a=*/false, /*reuse_b=*/false);
}

// ---------------------------------------------------------------------------
// fp32 -> bf16 bulk convert (4 elems/thread via float4)
// ---------------------------------------------------------------------------
__global__ void k_f32_to_bf16(const float* __restrict__ x, u16* __restrict__ y) {
    size_t i = ((size_t)blockIdx.x * blockDim.x + threadIdx.x) * 4;
    float4 v = *reinterpret_cast<const float4*>(x + i);
    y[i + 0] = f2bf(v.x); y[i + 1] = f2bf(v.y);
    y[i + 2] = f2bf(v.z); y[i + 3] = f2bf(v.w);
}

// W [D,D] fp32 row-major -> Wt [N=D rows, K=D cols] bf16 (i.e. W transposed)
__global__ void k_transpose_w(const float* __restrict__ W, u16* __restrict__ Wt) {
    size_t i = (size_t)blockIdx.x * blockDim.x + threadIdx.x;   // i = n*D + k
    int n = (int)(i >> 10);
    int k = (int)(i & (D_ - 1));
    Wt[i] = f2bf(W[(size_t)k * D_ + n]);
}

// ---------------------------------------------------------------------------
// Tiled bf16 GEMM: C[M,N] = A[M,K] * B[K,N], B given as Bt = B^T [N,K].
// Block tile 128x128, K-step 32, 8 waves in a 4x2 grid, each wave 32x64
// (2x4 grid of 16x16x32 WMMA accumulators). Tiles staged by the Tensor Data
// Mover (double-buffered, one issuing wave, s_wait_tensorcnt pipelining) when
// available, else by per-thread b128 loads.
// mode 0: fp32 store to outf[M,N]
// mode 1: bf16 store, head layout [B,H,S,dk]
// mode 2: mode 1 + RoPE (for Q/K)
// ---------------------------------------------------------------------------
__global__ void k_gemm_bf16(const u16* __restrict__ A, const u16* __restrict__ Bt,
                            int K, int mode,
                            float* __restrict__ outf, u16* __restrict__ outh,
                            const int* __restrict__ tpos) {
    __shared__ __align__(16) u16 As[2][128 * 32];
    __shared__ __align__(16) u16 Bs[2][128 * 32];

    const int t    = threadIdx.x;
    const int lane = t & 31;
    const int w    = t >> 5;
    const int half = lane >> 4;
    const int lcol = lane & 15;
    const int wr   = w >> 1;   // 0..3
    const int wc   = w & 1;    // 0..1
    const int bm0  = blockIdx.x * 128;
    const int bn0  = blockIdx.y * 128;

    v8f acc[2][4];
#pragma unroll
    for (int mi = 0; mi < 2; ++mi)
#pragma unroll
        for (int ni = 0; ni < 4; ++ni) acc[mi][ni] = (v8f)(0.0f);

    int buf = 0;

#if USE_TDM
    // ---- TDM double-buffered staging ------------------------------------
    const u16* Ag = A + (size_t)bm0 * K;
    const u16* Bg = Bt + (size_t)bn0 * K;
    if (w == 0) {
        tdm_load_2d(Ag, (unsigned)(size_t)&As[0][0], 32, 128, K, 128);
        tdm_load_2d(Bg, (unsigned)(size_t)&Bs[0][0], 32, 128, K, 128);
    }
    for (int k0 = 0; k0 < K; k0 += 32) {
        if (w == 0) {
            if (k0 + 32 < K) {
                tdm_load_2d(Ag + k0 + 32, (unsigned)(size_t)&As[buf ^ 1][0], 32, 128, K, 128);
                tdm_load_2d(Bg + k0 + 32, (unsigned)(size_t)&Bs[buf ^ 1][0], 32, 128, K, 128);
                TDM_WAIT(2);   // retire current pair (in-order), keep next in flight
            } else {
                TDM_WAIT(0);
            }
        }
        __syncthreads();

        Frag af[2], bf[4];
#pragma unroll
        for (int mi = 0; mi < 2; ++mi)
            af[mi] = load_frag(&As[buf][(wr * 32 + mi * 16) * 32], 32, 0, lane);
#pragma unroll
        for (int ni = 0; ni < 4; ++ni)
            bf[ni] = load_frag(&Bs[buf][(wc * 64 + ni * 16) * 32], 32, 0, lane);

#pragma unroll
        for (int mi = 0; mi < 2; ++mi)
#pragma unroll
            for (int ni = 0; ni < 4; ++ni)
                acc[mi][ni] = wmma_bf16(af[mi], bf[ni], acc[mi][ni]);
        __syncthreads();
        buf ^= 1;
    }
#else
    // ---- fallback: per-thread b128 staging ------------------------------
    const int lr = t >> 1;            // 0..127 (tile row this thread loads)
    const int lc = (t & 1) * 16;      // bf16 col offset 0/16
    for (int k0 = 0; k0 < K; k0 += 32) {
        const uint4* gA = reinterpret_cast<const uint4*>(A + (size_t)(bm0 + lr) * K + k0 + lc);
        const uint4* gB = reinterpret_cast<const uint4*>(Bt + (size_t)(bn0 + lr) * K + k0 + lc);
        uint4 a0 = gA[0], a1 = gA[1];
        uint4 b0 = gB[0], b1 = gB[1];
        if (k0 + 32 < K) {
            __builtin_prefetch(A + (size_t)(bm0 + lr) * K + k0 + 32 + lc, 0, 1);
            __builtin_prefetch(Bt + (size_t)(bn0 + lr) * K + k0 + 32 + lc, 0, 1);
        }
        uint4* sA = reinterpret_cast<uint4*>(&As[0][lr * 32 + lc]);
        uint4* sB = reinterpret_cast<uint4*>(&Bs[0][lr * 32 + lc]);
        sA[0] = a0; sA[1] = a1;
        sB[0] = b0; sB[1] = b1;
        __syncthreads();

        Frag af[2], bf[4];
#pragma unroll
        for (int mi = 0; mi < 2; ++mi)
            af[mi] = load_frag(&As[0][(wr * 32 + mi * 16) * 32], 32, 0, lane);
#pragma unroll
        for (int ni = 0; ni < 4; ++ni)
            bf[ni] = load_frag(&Bs[0][(wc * 64 + ni * 16) * 32], 32, 0, lane);

#pragma unroll
        for (int mi = 0; mi < 2; ++mi)
#pragma unroll
            for (int ni = 0; ni < 4; ++ni)
                acc[mi][ni] = wmma_bf16(af[mi], bf[ni], acc[mi][ni]);
        __syncthreads();
    }
#endif

    // --- epilogue --------------------------------------------------------
#pragma unroll
    for (int mi = 0; mi < 2; ++mi) {
#pragma unroll
        for (int ni = 0; ni < 4; ++ni) {
#pragma unroll
            for (int r = 0; r < 8; ++r) {
                float val = acc[mi][ni][r];
                int row = bm0 + wr * 32 + mi * 16 + r + 8 * half;   // global M
                int col = bn0 + wc * 64 + ni * 16 + lcol;           // global N
                if (mode == 0) {
                    outf[(size_t)row * D_ + col] = val;
                } else {
                    int b = row >> 11;          // row / S
                    int s = row & (S_ - 1);
                    int h = col >> 6;           // col / dk
                    int d = col & (DK_ - 1);
                    if (mode == 2) {
                        // RoPE: pair partner (col^1) lives in the adjacent lane
                        float partner = __shfl_xor(val, 1, 32);
                        int   i2   = d & ~1;
                        float freq = __expf(-((float)i2 * (1.0f / 64.0f)) * 9.210340372f);
                        float ang  = (float)tpos[s] * freq;
                        float sn, cs;
                        __sincosf(ang, &sn, &cs);
                        val = val * cs + ((d & 1) ? partner * sn : -partner * sn);
                    }
                    outh[(((size_t)(b * H_ + h)) * S_ + s) * DK_ + d] = f2bf(val);
                }
            }
        }
    }
}

// ---------------------------------------------------------------------------
// Flash attention, causal. One block per (128 query rows, b*H+h).
// 8 waves, each owns 16 query rows. K blocks staged by TDM (row-major); V
// staged transposed by per-thread scatter. Scores and P*V on
// v_wmma_f32_16x16x32_bf16; fp32 online softmax with shfl_xor reductions.
// ---------------------------------------------------------------------------
__global__ void k_flash_attn(const u16* __restrict__ Qb, const u16* __restrict__ Kb,
                             const u16* __restrict__ Vb, u16* __restrict__ Ab) {
    __shared__ __align__(16) u16 Ks[128 * 64];     // [key][dk]
    __shared__ __align__(16) u16 Vt[64 * 128];     // [dk][key] (transposed)
    __shared__ __align__(16) u16 Ps[8 * 16 * 32];  // per-wave P tile

    const int t    = threadIdx.x;
    const int lane = t & 31;
    const int w    = t >> 5;
    const int half = lane >> 4;
    const int lcol = lane & 15;
    const int bh   = blockIdx.y;
    const int b    = bh >> 4;
    const int h    = bh & 15;
    const int q0   = blockIdx.x * 128;
    const size_t head_off = (size_t)bh * S_ * DK_;

    // Q fragments: rows q0 + w*16 + lcol, two 16x32 chunks of dk=64.
    Frag qf[2];
    {
        const u16* qrow = Qb + head_off + (size_t)(q0 + w * 16 + lcol) * DK_;
#pragma unroll
        for (int kk = 0; kk < 2; ++kk) {
            const u16* p = qrow + kk * 32 + 8 * half;
            qf[kk].q[0] = *reinterpret_cast<const uint4*>(p);
            qf[kk].q[1] = *reinterpret_cast<const uint4*>(p + 16);
        }
    }

    v8f o[4];
#pragma unroll
    for (int ni = 0; ni < 4; ++ni) o[ni] = (v8f)(0.0f);
    float rm[8], rl[8];
#pragma unroll
    for (int r = 0; r < 8; ++r) { rm[r] = -1e30f; rl[r] = 0.0f; }

    const float sc = 0.125f;   // 1/sqrt(64)
    u16* pw = &Ps[w * 16 * 32];

    for (int jb = 0; jb <= blockIdx.x; ++jb) {
        const int j0 = jb * 128;
        // --- stage K (TDM row-major) and V (transposed scatter) ----------
#if USE_TDM
        if (w == 0)
            tdm_load_2d(Kb + head_off + (size_t)j0 * DK_,
                        (unsigned)(size_t)&Ks[0], 64, 128, 64, 128);
#endif
        {
            const int r_ = t >> 1;
            const int c_ = (t & 1) * 32;   // bf16 offset in the 64-wide row
#if !USE_TDM
            const uint4* gk = reinterpret_cast<const uint4*>(Kb + head_off + (size_t)(j0 + r_) * DK_ + c_);
            uint4* sk = reinterpret_cast<uint4*>(&Ks[r_ * 64 + c_]);
            sk[0] = gk[0]; sk[1] = gk[1]; sk[2] = gk[2]; sk[3] = gk[3];
#endif
            const uint4* gv = reinterpret_cast<const uint4*>(Vb + head_off + (size_t)(j0 + r_) * DK_ + c_);
#pragma unroll
            for (int u = 0; u < 4; ++u) {
                uint4 d4 = gv[u];
                const u16* e = reinterpret_cast<const u16*>(&d4);
#pragma unroll
                for (int x = 0; x < 8; ++x)
                    Vt[(c_ + u * 8 + x) * 128 + r_] = e[x];
            }
        }
#if USE_TDM
        if (w == 0) TDM_WAIT(0);
#endif
        __syncthreads();

        for (int nt2 = 0; nt2 < 4; ++nt2) {
            const int ntA = nt2 * 2, ntB = ntA + 1;
            // --- scores: two 16x16 tiles over 32 keys --------------------
            v8f s0 = (v8f)(0.0f), s1 = (v8f)(0.0f);
            s0 = wmma_bf16(qf[0], load_frag(&Ks[ntA * 16 * 64], 64, 0,  lane), s0);
            s0 = wmma_bf16(qf[1], load_frag(&Ks[ntA * 16 * 64], 64, 32, lane), s0);
            s1 = wmma_bf16(qf[0], load_frag(&Ks[ntB * 16 * 64], 64, 0,  lane), s1);
            s1 = wmma_bf16(qf[1], load_frag(&Ks[ntB * 16 * 64], 64, 32, lane), s1);

            const int colA = j0 + ntA * 16 + lcol;
            const int colB = j0 + ntB * 16 + lcol;
#pragma unroll
            for (int r = 0; r < 8; ++r) {
                const int row = q0 + w * 16 + r + 8 * half;   // global query idx
                float a  = s0[r] * sc; if (colA > row) a  = -1e30f;
                float bv = s1[r] * sc; if (colB > row) bv = -1e30f;
                // row max across the 16 lanes of this half
                float tm = fmaxf(a, bv);
#pragma unroll
                for (int msk = 1; msk < 16; msk <<= 1)
                    tm = fmaxf(tm, __shfl_xor(tm, msk, 32));
                float mnew = fmaxf(rm[r], tm);
                float corr = __expf(rm[r] - mnew);
                float pa = __expf(a - mnew);
                float pb = __expf(bv - mnew);
                float rs = pa + pb;
#pragma unroll
                for (int msk = 1; msk < 16; msk <<= 1)
                    rs += __shfl_xor(rs, msk, 32);
                rl[r] = rl[r] * corr + rs;
                rm[r] = mnew;
#pragma unroll
                for (int ni = 0; ni < 4; ++ni) o[ni][r] *= corr;
                // spill P to per-wave LDS in (row, key) order for A-repack
                pw[(r + 8 * half) * 32 + lcol]      = f2bf(pa);
                pw[(r + 8 * half) * 32 + 16 + lcol] = f2bf(pb);
            }
            asm volatile("s_wait_dscnt 0" ::: "memory");   // wave-local P visible
            Frag pf = load_frag(pw, 32, 0, lane);
#pragma unroll
            for (int ni = 0; ni < 4; ++ni) {
                Frag vf = load_frag(&Vt[(ni * 16) * 128], 128, nt2 * 32, lane);
                o[ni] = wmma_bf16(pf, vf, o[ni]);
            }
        }
        __syncthreads();
    }

    // --- normalize and store to [B,S,D] bf16 -----------------------------
#pragma unroll
    for (int ni = 0; ni < 4; ++ni) {
#pragma unroll
        for (int r = 0; r < 8; ++r) {
            float val = o[ni][r] / rl[r];
            int s = q0 + w * 16 + r + 8 * half;
            int d = ni * 16 + lcol;
            Ab[((size_t)b * S_ + s) * D_ + h * DK_ + d] = f2bf(val);
        }
    }
}

// ---------------------------------------------------------------------------
// Host-side launch
// ---------------------------------------------------------------------------
extern "C" void kernel_launch(void* const* d_in, const int* in_sizes, int n_in,
                              void* d_out, int out_size, void* d_ws, size_t ws_size,
                              hipStream_t stream) {
    const float* x  = (const float*)d_in[0];
    const float* Wq = (const float*)d_in[1];
    const float* Wk = (const float*)d_in[2];
    const float* Wv = (const float*)d_in[3];
    const float* Wo = (const float*)d_in[4];
    const int*   tp = (const int*)d_in[5];

    char* ws = (char*)d_ws;
    const size_t XB_BYTES = (size_t)M_ * D_ * 2;   // 16 MiB
    const size_t W_BYTES  = (size_t)D_ * D_ * 2;   //  2 MiB
    u16* Xb  = (u16*)(ws);
    u16* Wqb = (u16*)(ws + XB_BYTES);
    u16* Wkb = (u16*)(ws + XB_BYTES + 1 * W_BYTES);
    u16* Wvb = (u16*)(ws + XB_BYTES + 2 * W_BYTES);
    u16* Wob = (u16*)(ws + XB_BYTES + 3 * W_BYTES);
    u16* Qb  = (u16*)(ws + XB_BYTES + 4 * W_BYTES);
    u16* Kb  = (u16*)(ws + 2 * XB_BYTES + 4 * W_BYTES);
    u16* Vb  = (u16*)(ws + 3 * XB_BYTES + 4 * W_BYTES);
    u16* Ab  = (u16*)(ws + 4 * XB_BYTES + 4 * W_BYTES);
    float* out = (float*)d_out;

    // 1) convert x to bf16; transpose+convert weights
    k_f32_to_bf16<<<(M_ * D_) / (256 * 4), 256, 0, stream>>>(x, Xb);
    k_transpose_w<<<(D_ * D_) / 256, 256, 0, stream>>>(Wq, Wqb);
    k_transpose_w<<<(D_ * D_) / 256, 256, 0, stream>>>(Wk, Wkb);
    k_transpose_w<<<(D_ * D_) / 256, 256, 0, stream>>>(Wv, Wvb);
    k_transpose_w<<<(D_ * D_) / 256, 256, 0, stream>>>(Wo, Wob);

    // 2) QKV projections (+RoPE for Q/K), bf16 head-layout outputs
    dim3 gGemm(M_ / 128, D_ / 128);
    k_gemm_bf16<<<gGemm, 256, 0, stream>>>(Xb, Wqb, D_, 2, nullptr, Qb, tp);
    k_gemm_bf16<<<gGemm, 256, 0, stream>>>(Xb, Wkb, D_, 2, nullptr, Kb, tp);
    k_gemm_bf16<<<gGemm, 256, 0, stream>>>(Xb, Wvb, D_, 1, nullptr, Vb, tp);

    // 3) causal flash attention per (b,h)
    dim3 gAttn(S_ / 128, B_ * H_);
    k_flash_attn<<<gAttn, 256, 0, stream>>>(Qb, Kb, Vb, Ab);

    // 4) output projection, fp32 result
    k_gemm_bf16<<<gGemm, 256, 0, stream>>>(Ab, Wob, D_, 0, out, nullptr, tp);
}
